// GraphSAGELayer_lstm_22565758173858
// MI455X (gfx1250) — compile-verified, matrix operand order checked by reading, not run on hardware
//
#include <hip/hip_runtime.h>

typedef __bf16 bf16;
typedef __attribute__((ext_vector_type(16))) __bf16 v16bf;
typedef __attribute__((ext_vector_type(8)))  __bf16 v8bf;
typedef __attribute__((ext_vector_type(8)))  float  v8f;
typedef __attribute__((ext_vector_type(4)))  unsigned int u32x4;
typedef __attribute__((ext_vector_type(8)))  int   i32x8;
typedef __attribute__((ext_vector_type(4)))  int   i32x4;

#define NNODES 32768
#define DEG    32
#define FD     128      // in feats == hidden
#define HD     128
#define G4     512      // 4*H
#define TILE_M 16

#if __has_builtin(__builtin_amdgcn_tensor_load_to_lds)
#define USE_TDM 1
#else
#define USE_TDM 0
#endif

// LDS layout (dynamic shared, offset 0 = block LDS base)
#define LDS_OFF_H    0                        // [16][128] bf16  (4 KB)
#define LDS_OFF_XBF  4096                     // [16][128] bf16  (4 KB)
#define LDS_OFF_STG  8192                     // [2][16][128] f32 (16 KB)
#define LDS_BYTES    24576

// ---------------- prep: f32 -> bf16 weights, fused biases, transposed u16 indices ----
__global__ __launch_bounds__(256) void sage_prep_kernel(
    const float* __restrict__ Wih, const float* __restrict__ Whh,
    const float* __restrict__ bih, const float* __restrict__ bhh,
    const float* __restrict__ W1,  const float* __restrict__ b1,
    const float* __restrict__ W2,  const float* __restrict__ b2,
    const int* __restrict__ nidx,
    unsigned short* __restrict__ nidx16T,
    bf16* __restrict__ wWih, bf16* __restrict__ wWhh,
    bf16* __restrict__ wW1,  bf16* __restrict__ wW2,
    float* __restrict__ wb,  float* __restrict__ wb12)
{
    int tid = blockIdx.x * blockDim.x + threadIdx.x;
    if (tid < NNODES * DEG) {
        int n = tid >> 5;          // DEG == 32
        int d = tid & 31;
        nidx16T[d * NNODES + n] = (unsigned short)nidx[tid];  // [t][node], N fits u16
    }
    if (tid < G4 * FD) {
        wWih[tid] = (bf16)Wih[tid];
        wWhh[tid] = (bf16)Whh[tid];
    }
    if (tid < FD * FD) {
        wW1[tid] = (bf16)W1[tid];
        wW2[tid] = (bf16)(W2[tid] * (1.0f / (float)DEG));  // fold 1/deg (exact: pow2)
    }
    if (tid < G4) wb[tid]  = bih[tid] + bhh[tid];
    if (tid < FD) wb12[tid] = b1[tid] + b2[tid];
}

// ---------------- fragment loader ----------------
// Tile row-major [r][128] bf16. A: r = M row; B: r = row n of W (= col of W^T).
// ISA 7.12.2 (16-bit 16x32): lanes 0-15: r=lane, K=k0+{0..7,16..23};
// lanes 16-31: r=lane-16, K=k0+{8..15,24..31} -> two contiguous 16B loads.
__device__ __forceinline__ v16bf load_frag16(const bf16* base, int lane, int k0)
{
    int r  = lane & 15;
    int kb = (lane < 16) ? 0 : 8;
    const bf16* p = base + r * FD + k0 + kb;
    v8bf lo = *reinterpret_cast<const v8bf*>(p);
    v8bf hi = *reinterpret_cast<const v8bf*>(p + 16);
    v16bf out;
#pragma unroll
    for (int i = 0; i < 8; ++i) { out[i] = lo[i]; out[8 + i] = hi[i]; }
    return out;
}

// ---- fast transcendentals: native V_TANH_F32 on CDNA5 (TRANS32 op) ----
#if __has_builtin(__builtin_amdgcn_tanhf)
__device__ __forceinline__ float fast_tanh(float x) { return __builtin_amdgcn_tanhf(x); }
#else
__device__ __forceinline__ float fast_tanh(float x) { return tanhf(x); }
#endif
__device__ __forceinline__ float sigf(float x) { return 0.5f + 0.5f * fast_tanh(0.5f * x); }

#if USE_TDM
// Issue one TDM gather: 16 rows of 128 f32 from feat, selected by 16-bit row
// indices, into LDS staging buffer. Descriptor per CDNA5 ISA.
// This toolchain exposes the 6-arg builtin: (g0, g1, g2, g3, extra_v8i, cpol).
__device__ __forceinline__ void tdm_gather16(const float* feat,
                                             const unsigned short* nidx16T,
                                             int nodeBase, int t, int buf)
{
    unsigned long long ga = (unsigned long long)(size_t)feat;
    u32x4 g0;
    g0[0] = 1u | (1u << 31);                     // count=1, gather_mode=1, idx=16-bit
    g0[1] = (unsigned)(LDS_OFF_STG + buf * 8192);// lds_addr (bytes)
    g0[2] = (unsigned)(ga & 0xffffffffu);        // global_addr[31:0]
    g0[3] = (unsigned)((ga >> 32) & 0x1ffffffu)  // global_addr[56:32]
          | (2u << 30);                          // type=2 (image)
    i32x8 g1;
    g1[0] = (int)(2u << 16);        // data_size=2 (4 bytes); no multicast/iterate/pad
    g1[1] = (int)(128u << 16);      // tensor_dim0 = 128 (bits[79:48] low half)
    g1[2] = (int)(32768u << 16);    // tensor_dim1 = N  (bits[111:80] low half)
    g1[3] = (int)(128u << 16);      // tile_dim0 = 128 (bits[127:112])
    g1[4] = 16;                     // tile_dim1 = #valid indices = 16
    g1[5] = 128;                    // tensor_dim0_stride = 128 (low32)
    g1[6] = 0;
    g1[7] = 0;
    // groups 2/3: 16 contiguous u16 row indices (uniform address -> scalar loads)
    const i32x4* ip = reinterpret_cast<const i32x4*>(
        nidx16T + (size_t)t * NNODES + nodeBase);
    i32x4 g2 = ip[0];
    i32x4 g3 = ip[1];
    i32x8 gx;
#pragma unroll
    for (int i = 0; i < 8; ++i) gx[i] = 0;       // uncharacterized operand: zeros
    __builtin_amdgcn_tensor_load_to_lds(g0, g1, g2, g3, gx, 0);
}
#endif

// ---------------- main fused kernel ----------------
// One block (8 waves, wave32) per 16-node tile. Weights live in VGPRs:
// 32 loop-invariant B fragments per wave (256 VGPRs, via CDNA5 VGPR-MSB file).
// LDS carries only the per-step x tile (TDM gather target) and recurrent h tile.
__global__ __launch_bounds__(256) void sage_lstm_kernel(
    const float* __restrict__ feat, const int* __restrict__ nidx,
    const unsigned short* __restrict__ nidx16T,
    const bf16* __restrict__ wWih, const bf16* __restrict__ wWhh,
    const bf16* __restrict__ wW1,  const bf16* __restrict__ wW2,
    const float* __restrict__ wb,  const float* __restrict__ wb12,
    float* __restrict__ out)
{
    extern __shared__ char smem[];
    bf16*  sH   = reinterpret_cast<bf16*>(smem + LDS_OFF_H);    // [16][128]
    bf16*  sX   = reinterpret_cast<bf16*>(smem + LDS_OFF_XBF);  // [16][128]
    float* sStg = reinterpret_cast<float*>(smem + LDS_OFF_STG); // [2][2048]
    (void)sStg; (void)nidx; (void)nidx16T;

    const int tid    = threadIdx.x;
    const int lane   = tid & 31;
    const int wv     = tid >> 5;         // 0..7
    const int ln     = lane & 15;
    const int hiHalf = lane >> 4;        // C-layout: M = e + 8*hiHalf
    const int unit0  = wv * 16;          // this wave's 16 hidden units / out cols
    const int nodeBase = blockIdx.x * TILE_M;

    // ---- loop-invariant weight fragments -> registers (one-time, L2-resident)
    v16bf fWih[4][4];   // [gate][kchunk]
    v16bf fWhh[4][4];
#pragma unroll
    for (int g = 0; g < 4; ++g) {
        const bf16* baseI = wWih + (g * HD + unit0) * FD;
        const bf16* baseH = wWhh + (g * HD + unit0) * FD;
#pragma unroll
        for (int kc = 0; kc < 4; ++kc) {
            fWih[g][kc] = load_frag16(baseI, lane, kc * 32);
            fWhh[g][kc] = load_frag16(baseH, lane, kc * 32);
        }
    }

    // h0 = 0
    for (int i = tid; i < TILE_M * FD; i += 256) sH[i] = (bf16)0.0f;

    float bias[4];
#pragma unroll
    for (int g = 0; g < 4; ++g) bias[g] = wb[g * HD + unit0 + ln];

    v8f c;
#pragma unroll
    for (int e = 0; e < 8; ++e) c[e] = 0.0f;

    const int srow = tid >> 4;
    const int scol = (tid & 15) * 8;

#if USE_TDM
    if (wv == 0) tdm_gather16(feat, nidx16T, nodeBase, 0, 0);  // prefetch t=0
#endif

    for (int t = 0; t < DEG; ++t) {
        const int buf = t & 1;

#if USE_TDM
        if (wv == 0) {
            if (t + 1 < DEG) {
                tdm_gather16(feat, nidx16T, nodeBase, t + 1, buf ^ 1);  // async next
                __builtin_amdgcn_s_wait_tensorcnt(1);  // in-order: tile t landed
            } else {
                __builtin_amdgcn_s_wait_tensorcnt(0);
            }
        }
        __syncthreads();  // staging[buf] visible to all waves; prev sH stores done
        // f32 staging -> bf16 x tile (256 threads x 8 elems)
        {
            const float4* sp = reinterpret_cast<const float4*>(sStg + buf * 2048 + tid * 8);
            float4 f0 = sp[0];
            float4 f1 = sp[1];
            v8bf vv;
            vv[0] = (bf16)f0.x; vv[1] = (bf16)f0.y; vv[2] = (bf16)f0.z; vv[3] = (bf16)f0.w;
            vv[4] = (bf16)f1.x; vv[5] = (bf16)f1.y; vv[6] = (bf16)f1.z; vv[7] = (bf16)f1.w;
            *reinterpret_cast<v8bf*>(sX + tid * 8) = vv;
        }
#else
        // fallback: cooperative gather + convert
        {
            int nb = nidx[(nodeBase + srow) * DEG + t];
            const float4* src = reinterpret_cast<const float4*>(feat + nb * FD + scol);
            float4 f0 = src[0];
            float4 f1 = src[1];
            v8bf vv;
            vv[0] = (bf16)f0.x; vv[1] = (bf16)f0.y; vv[2] = (bf16)f0.z; vv[3] = (bf16)f0.w;
            vv[4] = (bf16)f1.x; vv[5] = (bf16)f1.y; vv[6] = (bf16)f1.z; vv[7] = (bf16)f1.w;
            *reinterpret_cast<v8bf*>(sX + srow * FD + scol) = vv;
        }
        __syncthreads();  // keep the h-visibility barrier in fallback too
#endif
        __syncthreads();  // x tile ready

        // gates = x@Wih^T + h@Whh^T + b : B operands already in VGPRs
        v8f acc[4];
#pragma unroll
        for (int g = 0; g < 4; ++g) {
            float bg = bias[g];
#pragma unroll
            for (int e = 0; e < 8; ++e) acc[g][e] = bg;
        }

#pragma unroll
        for (int kc = 0; kc < 4; ++kc) {
            const int k0 = kc * 32;
            v16bf aX = load_frag16(sX, lane, k0);
            v16bf aH = load_frag16(sH, lane, k0);
#pragma unroll
            for (int g = 0; g < 4; ++g) {
                acc[g] = __builtin_amdgcn_wmma_f32_16x16x32_bf16(
                    false, aX, false, fWih[g][kc], (short)0, acc[g], false, false);
                acc[g] = __builtin_amdgcn_wmma_f32_16x16x32_bf16(
                    false, aH, false, fWhh[g][kc], (short)0, acc[g], false, false);
            }
        }
        __syncthreads();  // all sX/sH reads done before overwrite

        // LSTM cell update in C-layout registers; h back to LDS as bf16.
        // sigmoid/tanh via native v_tanh_f32 (5 TRANS ops per element).
#pragma unroll
        for (int e = 0; e < 8; ++e) {
            float ig = sigf(acc[0][e]);
            float fg = sigf(acc[1][e]);
            float gg = fast_tanh(acc[2][e]);
            float og = sigf(acc[3][e]);
            float cn = fg * c[e] + ig * gg;
            c[e] = cn;
            float hn = og * fast_tanh(cn);
            int m = e + 8 * hiHalf;
            sH[m * FD + unit0 + ln] = (bf16)hn;
        }
    }

    // stage own feat rows for the output GEMM
    {
        const float4* src = reinterpret_cast<const float4*>(feat + (nodeBase + srow) * FD + scol);
        float4 f0 = src[0];
        float4 f1 = src[1];
        v8bf vv;
        vv[0] = (bf16)f0.x; vv[1] = (bf16)f0.y; vv[2] = (bf16)f0.z; vv[3] = (bf16)f0.w;
        vv[4] = (bf16)f1.x; vv[5] = (bf16)f1.y; vv[6] = (bf16)f1.z; vv[7] = (bf16)f1.w;
        *reinterpret_cast<v8bf*>(sX + srow * FD + scol) = vv;
    }
    __syncthreads();  // final h + feat tile visible

    // out = feat@W1^T + h@(W2/deg)^T + (b1+b2); wave w -> out cols [16w,16w+16)
    float bo = wb12[unit0 + ln];
    v8f oacc;
#pragma unroll
    for (int e = 0; e < 8; ++e) oacc[e] = bo;

#pragma unroll
    for (int kc = 0; kc < 4; ++kc) {
        const int k0 = kc * 32;
        v16bf aF = load_frag16(sX, lane, k0);
        v16bf aH = load_frag16(sH, lane, k0);
        v16bf b1f = load_frag16(wW1 + unit0 * FD, lane, k0);
        oacc = __builtin_amdgcn_wmma_f32_16x16x32_bf16(
            false, aF, false, b1f, (short)0, oacc, false, false);
        v16bf b2f = load_frag16(wW2 + unit0 * FD, lane, k0);
        oacc = __builtin_amdgcn_wmma_f32_16x16x32_bf16(
            false, aH, false, b2f, (short)0, oacc, false, false);
    }

#pragma unroll
    for (int e = 0; e < 8; ++e) {
        int m = e + 8 * hiHalf;
        out[(nodeBase + m) * FD + unit0 + ln] = oacc[e];
    }
}

// ---------------- launch ----------------
extern "C" void kernel_launch(void* const* d_in, const int* in_sizes, int n_in,
                              void* d_out, int out_size, void* d_ws, size_t ws_size,
                              hipStream_t stream)
{
    const float* feat = (const float*)d_in[0];
    const int*   nidx = (const int*)d_in[1];
    const float* Wih  = (const float*)d_in[2];
    const float* Whh  = (const float*)d_in[3];
    const float* bih  = (const float*)d_in[4];
    const float* bhh  = (const float*)d_in[5];
    const float* W1   = (const float*)d_in[6];
    const float* b1   = (const float*)d_in[7];
    const float* W2   = (const float*)d_in[8];
    const float* b2   = (const float*)d_in[9];
    float* out = (float*)d_out;

    // workspace: transposed u16 indices (2MB) + bf16 weights + fused biases
    unsigned short* nidx16T = (unsigned short*)d_ws;
    bf16*  wWih = (bf16*)(nidx16T + (size_t)NNODES * DEG);
    bf16*  wWhh = wWih + G4 * FD;
    bf16*  wW1  = wWhh + G4 * FD;
    bf16*  wW2  = wW1 + FD * FD;
    float* wb   = (float*)(wW2 + FD * FD);
    float* wb12 = wb + G4;

    sage_prep_kernel<<<(NNODES * DEG + 255) / 256, 256, 0, stream>>>(
        Wih, Whh, bih, bhh, W1, b1, W2, b2, nidx, nidx16T,
        wWih, wWhh, wW1, wW2, wb, wb12);

    sage_lstm_kernel<<<NNODES / TILE_M, 256, LDS_BYTES, stream>>>(
        feat, nidx, nidx16T, wWih, wWhh, wW1, wW2, wb, wb12, out);
}
